// Memory_88656714925588
// MI455X (gfx1250) — compile-verified, matrix operand order checked by reading, not run on hardware
//
#include <hip/hip_runtime.h>
#include <hip/hip_bf16.h>
#include <math.h>

typedef __attribute__((ext_vector_type(16))) __bf16 v16bf;
typedef __attribute__((ext_vector_type(8)))  __bf16 v8bf;
typedef __attribute__((ext_vector_type(8)))  float  v8f;
typedef __attribute__((ext_vector_type(4)))  unsigned int u32x4;
typedef __attribute__((ext_vector_type(8)))  int          i32x8;
typedef __attribute__((ext_vector_type(4)))  int          i32x4;

#define CK      64
#define NE      32768
#define Q       4096
#define CV      256
#define TOPK    20
#define NSLICE  16
#define MSLICE  (NE / NSLICE)      // 2048 memory rows per block slice
#define CHUNK_M 128                // rows of mem_k staged in LDS per chunk
#define NCHUNK  (MSLICE / CHUNK_M) // 16
#define QGROUP  256                // queries per block (8 waves x 32)
#define NQG     (Q / QGROUP)       // 16
#define LDA     72                 // padded LDS row (bf16) to avoid bank conflicts

#define NEG_INF (-__builtin_inff())

#if defined(__HIP_DEVICE_COMPILE__) && defined(__has_builtin)
#if __has_builtin(__builtin_amdgcn_tensor_load_to_lds)
#define HAVE_TDM 1
#endif
#endif
#ifndef HAVE_TDM
#define HAVE_TDM 0
#endif

// Dynamic-LDS layout for match_kernel (bytes)
#define F_OFF     0                         // f32 TDM tile [CK][CHUNK_M] = 32768
#define A_OFF     32768                     // bf16 [CHUNK_M][LDA]        = 18432
#define KSQ_OFF   51200                     // f32 [CHUNK_M]              = 512
#define MINK_OFF  51712                     // f32 [8]                    = 32
#define STAGE_OFF 51744                     // f32 [8][16*32]             = 16384
#define TOPV_OFF  68128                     // f32 [8][32][TOPK]          = 20480
#define TOPI_OFF  88608                     // i32 [8][32][TOPK]          = 20480
#define LDS_TOTAL 109088

// ---------------------------------------------------------------------------
// Wave helpers (wave32)
// ---------------------------------------------------------------------------
__device__ inline float wave_min_f(float v) {
#pragma unroll
    for (int m = 16; m > 0; m >>= 1) v = fminf(v, __shfl_xor(v, m, 32));
    return v;
}
__device__ inline void wave_lds_sync() {
    // LDS ops from one wave execute in order; just stop compiler reordering.
    __builtin_amdgcn_wave_barrier();
    asm volatile("" ::: "memory");
}

// ---------------------------------------------------------------------------
// Tensor Data Mover: DMA mem_k tile [CK rows x CHUNK_M contiguous] -> LDS f32.
// D# per CDNA5 ISA ch.8: Group0 {count|lds_addr|global_addr|type=2},
// Group1 {data_size=4B, tensor dims/strides, tile dims}. 2D -> groups 2/3 zero.
// ---------------------------------------------------------------------------
__device__ inline void tdm_load_tile(unsigned int lds_off, const float* gsrc) {
#if HAVE_TDM
    unsigned long long ga = (unsigned long long)(uintptr_t)gsrc;
    u32x4 g0;
    g0.x = 0x1u;                                   // count=1, user descriptor
    g0.y = lds_off;                                // LDS byte address
    g0.z = (unsigned int)ga;                       // global_addr[31:0]
    g0.w = (unsigned int)((ga >> 32) & 0x01FFFFFFu) | (2u << 30);  // addr[56:32] | type=2
    i32x8 g1;
    g1[0] = (int)(2u << 16);                       // data_size = 4 bytes
    g1[1] = (int)(((unsigned)NE & 0xFFFFu) << 16); // tensor_dim0[15:0]
    g1[2] = (int)((((unsigned)NE >> 16) & 0xFFFFu) | ((unsigned)CK << 16)); // dim0 hi | dim1 lo
    g1[3] = (int)((unsigned)CHUNK_M << 16);        // dim1 hi=0 | tile_dim0=128
    g1[4] = (int)(unsigned)CK;                     // tile_dim1=64 | tile_dim2=0
    g1[5] = (int)(unsigned)NE;                     // tensor_dim0_stride lo32
    g1[6] = 0;                                     // stride hi | dim1_stride lo
    g1[7] = 0;
    i32x4 z4 = {0, 0, 0, 0};
#if __clang_major__ >= 23
    i32x8 z8 = {0, 0, 0, 0, 0, 0, 0, 0};
    __builtin_amdgcn_tensor_load_to_lds(g0, g1, z4, z4, z8, 0);
#else
    __builtin_amdgcn_tensor_load_to_lds(g0, g1, z4, z4, 0);
#endif
#else
    (void)lds_off; (void)gsrc;
#endif
}

// ---------------------------------------------------------------------------
// Kernel 1: ksq[n] = sum_c mem_k[c][n]^2
// ---------------------------------------------------------------------------
__global__ void ksq_kernel(const float* __restrict__ mem_k, float* __restrict__ ksq) {
    int n = blockIdx.x * blockDim.x + threadIdx.x;
    float s = 0.f;
#pragma unroll 8
    for (int c = 0; c < CK; ++c) {
        float v = mem_k[c * NE + n];
        s += v * v;
    }
    ksq[n] = s;
}

// ---------------------------------------------------------------------------
// Kernels 2/3: transpose mem_v [CV, NE] -> [NE, CV]
// ---------------------------------------------------------------------------
__global__ void transpose_kernel(const float* __restrict__ src, float* __restrict__ dst) {
    __shared__ float tile[32][33];
    const int nb = blockIdx.x * 32;
    const int cb = blockIdx.y * 32;
    const int tx = threadIdx.x;
    const int ty = threadIdx.y;
#pragma unroll
    for (int i = 0; i < 32; i += 8)
        tile[ty + i][tx] = src[(size_t)(cb + ty + i) * NE + (nb + tx)];
    __syncthreads();
#pragma unroll
    for (int i = 0; i < 32; i += 8)
        dst[(size_t)(nb + ty + i) * CV + (cb + tx)] = tile[tx][ty + i];
}

// ---------------------------------------------------------------------------
// Kernel 4: fused affinity GEMM (bf16 WMMA) + TDM double-buffered mem_k
// staging + ballot-based tile early-out streaming top-20.
// Grid: NSLICE*NQG = 256 blocks of 256 threads (8 waves, 32 queries/wave).
// ---------------------------------------------------------------------------
__launch_bounds__(256, 1)
__global__ void match_kernel(const float* __restrict__ qk,
                             const float* __restrict__ mem_k,
                             const float* __restrict__ ksq,
                             float* __restrict__ part_val,
                             int*   __restrict__ part_idx) {
    extern __shared__ char smem[];
    float*  ldsF   = (float*) (smem + F_OFF);     // TDM dest: [c][m] f32
    __bf16* ldsA   = (__bf16*)(smem + A_OFF);     // [m][c] bf16, padded rows
    float*  ldsKsq = (float*) (smem + KSQ_OFF);
    float*  minKsq = (float*) (smem + MINK_OFF);

    const int tid   = threadIdx.x;
    const int wave  = tid >> 5;
    const int lane  = tid & 31;
    const int slice = blockIdx.x / NQG;
    const int qgrp  = blockIdx.x % NQG;
    const int mbase = slice * MSLICE;
    const int qbase = qgrp * QGROUP + wave * 32;

    const int nq   = lane & 15;
    const int half = lane >> 4;

    float* st = (float*)(smem + STAGE_OFF) + wave * (16 * 32);
    float* tv = (float*)(smem + TOPV_OFF) + (size_t)(wave * 32 + lane) * TOPK;
    int*   ti = (int*)  (smem + TOPI_OFF) + (size_t)(wave * 32 + lane) * TOPK;

#pragma unroll
    for (int k = 0; k < TOPK; ++k) { tv[k] = NEG_INF; ti[k] = 0; }

    // B fragments: two 16-query tiles per wave, both K=32 steps, kept in VGPRs.
    // Element e of step s maps to K = s*32 + 8*half + (e<8 ? e : e+8).
    v16bf bq[2][2];
#pragma unroll
    for (int t = 0; t < 2; ++t)
#pragma unroll
        for (int s = 0; s < 2; ++s)
#pragma unroll
            for (int e = 0; e < 16; ++e) {
                int k = s * 32 + 8 * half + ((e < 8) ? e : (e + 8));
                bq[t][s][e] = (__bf16)qk[k * Q + qbase + t * 16 + nq];
            }

    // A-fragment loader: two ds_load_b128 per K=32 step from the staged tile.
    auto load_afrag = [&](int sub, v16bf aq[2]) {
        const __bf16* arow = &ldsA[(sub * 16 + nq) * LDA];
#pragma unroll
        for (int s = 0; s < 2; ++s) {
            v8bf lo = *(const v8bf*)(arow + s * 32 + 8 * half);
            v8bf hi = *(const v8bf*)(arow + s * 32 + 8 * half + 16);
            aq[s] = __builtin_shufflevector(lo, hi,
                      0, 1, 2, 3, 4, 5, 6, 7, 8, 9, 10, 11, 12, 13, 14, 15);
        }
    };

    // ---------------- prologue: stage chunk 0 ----------------
#if HAVE_TDM
    if (tid == 0) {
        tdm_load_tile((unsigned int)(uintptr_t)ldsF, mem_k + mbase);
        __builtin_amdgcn_s_wait_tensorcnt(0);
    }
    __syncthreads();
    for (int i = tid; i < CHUNK_M * CK; i += 256) {
        int c = i >> 7, m = i & 127;
        ldsA[m * LDA + c] = (__bf16)ldsF[c * CHUNK_M + m];
    }
#else
    for (int i = tid; i < CHUNK_M * CK; i += 256) {
        int c = i >> 7, m = i & 127;
        ldsA[m * LDA + c] = (__bf16)mem_k[c * NE + mbase + m];
    }
#endif
    if (tid < CHUNK_M) ldsKsq[tid] = ksq[mbase + tid];
    if (tid < 8) {
        float mn = ksq[mbase + tid * 16];
        for (int j = 1; j < 16; ++j) mn = fminf(mn, ksq[mbase + tid * 16 + j]);
        minKsq[tid] = mn;
    }
    __syncthreads();

    float curmin = NEG_INF;

    for (int ch = 0; ch < NCHUNK; ++ch) {
        const int m0 = mbase + ch * CHUNK_M;
        const bool more = (ch + 1 < NCHUNK);

#if HAVE_TDM
        if (more && tid == 0)
            tdm_load_tile((unsigned int)(uintptr_t)ldsF, mem_k + m0 + CHUNK_M);
#endif
        // Conservative (stale) per-wave acceptance threshold for this chunk.
        const float thresh = wave_min_f(curmin);

        v16bf aqc[2];
        load_afrag(0, aqc);

        for (int sub = 0; sub < 8; ++sub) {
            const int msub = sub * 16;

            // Software pipeline: issue next sub's LDS loads before the WMMAs
            // so s_wait_dscnt hides under the matrix ops.
            v16bf aqn[2];
            if (sub < 7) load_afrag(sub + 1, aqn);

            v8f c0 = {}, c1 = {};
            c0 = __builtin_amdgcn_wmma_f32_16x16x32_bf16(false, aqc[0], false, bq[0][0],
                                                         (short)0, c0, false, false);
            c0 = __builtin_amdgcn_wmma_f32_16x16x32_bf16(false, aqc[1], false, bq[0][1],
                                                         (short)0, c0, false, false);
            c1 = __builtin_amdgcn_wmma_f32_16x16x32_bf16(false, aqc[0], false, bq[1][0],
                                                         (short)0, c1, false, false);
            c1 = __builtin_amdgcn_wmma_f32_16x16x32_bf16(false, aqc[1], false, bq[1][1],
                                                         (short)0, c1, false, false);

            // Tile early-out via ballot: wave_max(lmax) > thresh  <=>
            // any lane's local max exceeds the (wave-uniform) bound.
            float lmax = fmaxf(c0[0], c1[0]);
#pragma unroll
            for (int r = 1; r < 8; ++r) lmax = fmaxf(lmax, fmaxf(c0[r], c1[r]));
            const float affmax = (2.0f * lmax - minKsq[sub]) * 0.125f;

            if (__any(affmax > thresh)) {   // wave-uniform decision, no LDS traffic
#pragma unroll
                for (int r = 0; r < 8; ++r) {
                    st[(r + 8 * half) * 32 + nq]      = c0[r];
                    st[(r + 8 * half) * 32 + 16 + nq] = c1[r];
                }
                wave_lds_sync();
                // Every lane owns one query (column = lane).
                for (int i = 0; i < 16; ++i) {
                    float dot = st[i * 32 + lane];
                    float aff = (2.0f * dot - ldsKsq[msub + i]) * 0.125f; // 1/sqrt(64)
                    if (aff > curmin) {
                        int amin = 0; float vmin = tv[0];
#pragma unroll
                        for (int k = 1; k < TOPK; ++k)
                            if (tv[k] < vmin) { vmin = tv[k]; amin = k; }
                        tv[amin] = aff;
                        ti[amin] = m0 + msub + i;
                        vmin = tv[0];
#pragma unroll
                        for (int k = 1; k < TOPK; ++k) vmin = fminf(vmin, tv[k]);
                        curmin = vmin;
                    }
                }
                wave_lds_sync();
            }

            if (sub < 7) { aqc[0] = aqn[0]; aqc[1] = aqn[1]; }
        }

        if (more) {
            const int mn0 = m0 + CHUNK_M;
#if HAVE_TDM
            if (tid == 0) __builtin_amdgcn_s_wait_tensorcnt(0);
            __syncthreads();            // F ready; all waves done with A/ksq
            for (int i = tid; i < CHUNK_M * CK; i += 256) {
                int c = i >> 7, m = i & 127;
                ldsA[m * LDA + c] = (__bf16)ldsF[c * CHUNK_M + m];
            }
#else
            __syncthreads();
            for (int i = tid; i < CHUNK_M * CK; i += 256) {
                int c = i >> 7, m = i & 127;
                ldsA[m * LDA + c] = (__bf16)mem_k[c * NE + mn0 + m];
            }
#endif
            if (tid < CHUNK_M) ldsKsq[tid] = ksq[mn0 + tid];
            if (tid < 8) {
                float mn = ksq[mn0 + tid * 16];
                for (int j = 1; j < 16; ++j) mn = fminf(mn, ksq[mn0 + tid * 16 + j]);
                minKsq[tid] = mn;
            }
            __syncthreads();
        }
    }

    __syncthreads();
    {
        int q = qbase + lane;
#pragma unroll
        for (int k = 0; k < TOPK; ++k) {
            part_val[((size_t)slice * Q + q) * TOPK + k] = tv[k];
            part_idx[((size_t)slice * Q + q) * TOPK + k] = ti[k];
        }
    }
}

// ---------------------------------------------------------------------------
// Kernel 5: merge 16 partial top-20 lists -> exact top-20 + softmax weights.
// ---------------------------------------------------------------------------
__global__ void merge_topk_kernel(const float* __restrict__ part_val,
                                  const int*   __restrict__ part_idx,
                                  float* __restrict__ w_out,
                                  int*   __restrict__ idx_out) {
    int q = blockIdx.x * blockDim.x + threadIdx.x;
    if (q >= Q) return;

    float vals[NSLICE * TOPK];
    int   idxs[NSLICE * TOPK];
    for (int s = 0; s < NSLICE; ++s)
        for (int k = 0; k < TOPK; ++k) {
            vals[s * TOPK + k] = part_val[((size_t)s * Q + q) * TOPK + k];
            idxs[s * TOPK + k] = part_idx[((size_t)s * Q + q) * TOPK + k];
        }

    float selv[TOPK]; int seli[TOPK];
    for (int t = 0; t < TOPK; ++t) {
        int am = 0; float vm = vals[0];
        for (int j = 1; j < NSLICE * TOPK; ++j)
            if (vals[j] > vm) { vm = vals[j]; am = j; }
        selv[t] = vm; seli[t] = idxs[am];
        vals[am] = NEG_INF;
    }

    float mx = selv[0];
    float w[TOPK]; float sum = 0.f;
    for (int t = 0; t < TOPK; ++t) { w[t] = __expf(selv[t] - mx); sum += w[t]; }
    float inv = 1.0f / sum;
    for (int t = 0; t < TOPK; ++t) {
        w_out[(size_t)q * TOPK + t]   = w[t] * inv;
        idx_out[(size_t)q * TOPK + t] = seli[t];
    }
}

// ---------------------------------------------------------------------------
// Kernel 6: readout from transposed value banks; coalesced gathers + stores.
// ---------------------------------------------------------------------------
__launch_bounds__(256)
__global__ void readout_kernel(const float* __restrict__ v1T,
                               const float* __restrict__ v2T,
                               const float* __restrict__ w,
                               const int*   __restrict__ idx,
                               float* __restrict__ out) {
    __shared__ float sw[16][TOPK];
    __shared__ int   si[16][TOPK];
    __shared__ float s1[16][17];
    __shared__ float s2[16][17];

    const int tid = threadIdx.x;
    const int qb  = blockIdx.x * 16;
    const int c_l = tid & 15;   // minor -> coalesced gather reads
    const int q_l = tid >> 4;

    for (int i = tid; i < 16 * TOPK; i += 256) {
        int qq = i / TOPK, kk = i % TOPK;
        sw[qq][kk] = w[(size_t)(qb + qq) * TOPK + kk];
        si[qq][kk] = idx[(size_t)(qb + qq) * TOPK + kk];
    }
    __syncthreads();

    for (int cc = 0; cc < CV; cc += 16) {
        float a1 = 0.f, a2 = 0.f;
#pragma unroll 4
        for (int k = 0; k < TOPK; ++k) {
            int   n  = si[q_l][k];
            float wk = sw[q_l][k];
            a1 += wk * v1T[(size_t)n * CV + cc + c_l];
            a2 += wk * v2T[(size_t)n * CV + cc + c_l];
        }
        s1[q_l][c_l] = a1;
        s2[q_l][c_l] = a2;
        __syncthreads();
        int qm = tid & 15;
        int cm = tid >> 4;
        out[(size_t)(cc + cm) * Q + qb + qm]      = s1[qm][cm];
        out[(size_t)(CV + cc + cm) * Q + qb + qm] = s2[qm][cm];
        __syncthreads();
    }
}

// ---------------------------------------------------------------------------
// Launch wrapper
// ---------------------------------------------------------------------------
extern "C" void kernel_launch(void* const* d_in, const int* in_sizes, int n_in,
                              void* d_out, int out_size, void* d_ws, size_t ws_size,
                              hipStream_t stream) {
    const float* qk     = (const float*)d_in[0];
    const float* mem_k  = (const float*)d_in[1];
    const float* mem_v1 = (const float*)d_in[2];
    const float* mem_v2 = (const float*)d_in[3];
    // d_in[4] = top_k (20), baked in as TOPK.

    char* ws = (char*)d_ws;
    size_t off = 0;
    float* v1T      = (float*)(ws + off); off += (size_t)NE * CV * 4;
    float* v2T      = (float*)(ws + off); off += (size_t)NE * CV * 4;
    float* ksq      = (float*)(ws + off); off += (size_t)NE * 4;
    float* part_val = (float*)(ws + off); off += (size_t)NSLICE * Q * TOPK * 4;
    int*   part_idx = (int*)  (ws + off); off += (size_t)NSLICE * Q * TOPK * 4;
    float* wsW      = (float*)(ws + off); off += (size_t)Q * TOPK * 4;
    int*   wsIdx    = (int*)  (ws + off); off += (size_t)Q * TOPK * 4;

    float* out = (float*)d_out;

    ksq_kernel<<<NE / 256, 256, 0, stream>>>(mem_k, ksq);

    dim3 tgrid(NE / 32, CV / 32);
    dim3 tblk(32, 8);
    transpose_kernel<<<tgrid, tblk, 0, stream>>>(mem_v1, v1T);
    transpose_kernel<<<tgrid, tblk, 0, stream>>>(mem_v2, v2T);

    match_kernel<<<NSLICE * NQG, 256, LDS_TOTAL, stream>>>(qk, mem_k, ksq,
                                                           part_val, part_idx);

    merge_topk_kernel<<<Q / 256, 256, 0, stream>>>(part_val, part_idx, wsW, wsIdx);

    readout_kernel<<<Q / 16, 256, 0, stream>>>(v1T, v2T, wsW, wsIdx, out);
}
